// S6Layer_41764261987304
// MI455X (gfx1250) — compile-verified, matrix operand order checked by reading, not run on hardware
//
#include <hip/hip_runtime.h>
#include <hip/hip_bf16.h>
#include <math.h>

typedef __bf16 bf16;
typedef __bf16 v16bf  __attribute__((ext_vector_type(16)));
typedef __bf16 bf16x8 __attribute__((ext_vector_type(8)));
typedef float  v8f    __attribute__((ext_vector_type(8)));
typedef unsigned int u32x4 __attribute__((ext_vector_type(4)));
typedef int          i32x4 __attribute__((ext_vector_type(4)));
typedef int          i32x8 __attribute__((ext_vector_type(8)));

#define S6_B   8
#define S6_L   4096
#define S6_M   (S6_B * S6_L)     /* 32768 token rows */
#define S6_DM  256
#define S6_DI  384
#define S6_DS  8
#define S6_CH  32
#define S6_NC  (S6_L / S6_CH)    /* 128 chunks per sequence */

__device__ __forceinline__ float fast_sigmoid(float v) {
  return __builtin_amdgcn_rcpf(1.0f + __expf(-v));   // v_rcp_f32 + v_exp_f32
}

// ---------------- fp32 -> bf16 convert ----------------
__global__ __launch_bounds__(256)
void cvt_bf16_kernel(const float* __restrict__ in, bf16* __restrict__ out, int n) {
  int i = blockIdx.x * 256 + threadIdx.x;
  if (i < n) out[i] = (bf16)in[i];
}

// ---------------- LayerNorm (row = 256 elems, 1 block/row) ----------------
__global__ __launch_bounds__(256)
void ln_kernel(const float* __restrict__ x, const float* __restrict__ g,
               const float* __restrict__ b, bf16* __restrict__ xn) {
  __shared__ float red[16];
  const int row = blockIdx.x;
  const int tid = threadIdx.x;
  const int wid = tid >> 5, lane = tid & 31;

  float v = x[(long)row * S6_DM + tid];

  float s = v;
  #pragma unroll
  for (int off = 16; off > 0; off >>= 1) s += __shfl_xor(s, off, 32);
  if (lane == 0) red[wid] = s;
  __syncthreads();
  float tot = 0.f;
  #pragma unroll
  for (int w = 0; w < 8; ++w) tot += red[w];
  const float mu = tot * (1.0f / 256.0f);

  float dv = v - mu;
  float sq = dv * dv;
  #pragma unroll
  for (int off = 16; off > 0; off >>= 1) sq += __shfl_xor(sq, off, 32);
  if (lane == 0) red[8 + wid] = sq;
  __syncthreads();
  float vtot = 0.f;
  #pragma unroll
  for (int w = 0; w < 8; ++w) vtot += red[8 + w];
  const float rstd = rsqrtf(vtot * (1.0f / 256.0f) + 1e-5f);

  xn[(long)row * S6_DM + tid] = (bf16)(dv * rstd * g[tid] + b[tid]);
}

// ---------------- TDM: async 2-D bf16 tile (rows x 32) -> LDS --------------
// LDS layout produced: rows of 32 bf16 (64B = 16 DWORDs) padded by 8 DWORDs
// -> effective row stride 48 elements (96B), 16B aligned for ds_load_b128.
// D# bit packing per CDNA5 ISA ch.8 (group0: count/lds/global/type,
// group1: data_size=2B, pad_enable, pad_interval=16DW, pad_amount=8DW,
// tensor dims/strides, tile dims).
__device__ __forceinline__ void tdm_load_tile(unsigned lds_addr, const bf16* gptr,
                                              int row_len_elems, int tile_rows) {
  unsigned long long ga = (unsigned long long)(uintptr_t)gptr;
  u32x4 g0;
  g0[0] = 1u;                                              // count=1, is_restore=0
  g0[1] = lds_addr;                                        // lds_addr (bytes)
  g0[2] = (unsigned)(ga & 0xFFFFFFFFu);                    // global_addr lo
  g0[3] = (unsigned)((ga >> 32) & 0x1FFFFFFu) | (2u << 30);// global_addr hi | type=2
  i32x8 g1;
  g1[0] = (int)((1u << 16)   |   // data_size = 1 -> 2 bytes
                (1u << 20)   |   // pad_enable
                (3u << 22)   |   // pad_interval: 16 DWORDs written...
                (7u << 25));     // ...then +8 DWORDs pad -> 96B row stride
  g1[1] = (int)(((unsigned)row_len_elems & 0xFFFFu) << 16);            // tensor_dim0 lo16
  g1[2] = (int)((((unsigned)row_len_elems >> 16) & 0xFFFFu) |
                (((unsigned)tile_rows & 0xFFFFu) << 16));              // dim0 hi | dim1 lo
  g1[3] = (int)(32u << 16);                                            // dim1 hi=0 | tile_dim0=32
  g1[4] = (int)((unsigned)tile_rows & 0xFFFFu);                        // tile_dim1 | tile_dim2=0
  g1[5] = row_len_elems;                                               // tensor_dim0_stride lo32
  g1[6] = 0;
  g1[7] = 0;
  i32x4 z4 = {0, 0, 0, 0};
#if defined(__clang_major__) && (__clang_major__ >= 23)
  i32x8 z8 = {0, 0, 0, 0, 0, 0, 0, 0};
  __builtin_amdgcn_tensor_load_to_lds(g0, g1, z4, z4, z8, 0);
#else
  __builtin_amdgcn_tensor_load_to_lds(g0, g1, z4, z4, 0);
#endif
}

// ---------------- WMMA GEMM: C[M,N] = A[M,K] @ W[N,K]^T ----------------
// Block: 256 thr (8 waves). Block tile 128(M) x 128(N). Wave tile 64x32.
// TDM double-buffers A/B 128x32 bf16 tiles into LDS; waves consume with
// ds_load_b128 fragment loads matching the ISA 16-bit A/B VGPR layout.
// MODE 0: +b_in, silu/silu/raw split into 3 bf16 buffers of width DI
// MODE 1: +b_dt, softplus -> fp32 dt
// MODE 2: +b_out + residual -> fp32 out
template<int MODE>
__global__ __launch_bounds__(256)
void gemm_bf16_wmma(const bf16* __restrict__ A, const bf16* __restrict__ W,
                    const float* __restrict__ bias, int K, int N,
                    float* __restrict__ outf,
                    bf16* __restrict__ o_xp, bf16* __restrict__ o_z,
                    bf16* __restrict__ o_dt,
                    const float* __restrict__ resid) {
  constexpr int BM = 128, BN = 128, KT = 32, LD = 48;
  __shared__ __align__(16) bf16 As[2][BM * LD];
  __shared__ __align__(16) bf16 Bs[2][BN * LD];

  const int tid  = threadIdx.x;
  const int wid  = tid >> 5, lane = tid & 31;
  const int lr   = lane & 15, lh = lane >> 4;       // fragment row / K-half
  const int wm   = wid >> 2,  wn = wid & 3;         // wave grid 2 x 4
  const int bm   = blockIdx.x * BM;
  const int bn   = blockIdx.y * BN;
  const int nk   = K / KT;

  v8f c[4][2];
  const v8f vzero = {0.f,0.f,0.f,0.f,0.f,0.f,0.f,0.f};
  #pragma unroll
  for (int i = 0; i < 4; ++i)
    #pragma unroll
    for (int j = 0; j < 2; ++j) c[i][j] = vzero;

  // prologue: wave 0 launches TDM for K-tile 0 into buffer 0
  if (wid == 0) {
    tdm_load_tile((unsigned)(uintptr_t)&As[0][0], A + (long)bm * K, K, BM);
    tdm_load_tile((unsigned)(uintptr_t)&Bs[0][0], W + (long)bn * K, K, BN);
  }

  for (int kt = 0; kt < nk; ++kt) {
    const int cur = kt & 1;
    if (wid == 0) __builtin_amdgcn_s_wait_tensorcnt(0);
    __syncthreads();                                 // tiles[cur] visible to all

    if (wid == 0 && (kt + 1) < nk) {                 // prefetch next K-tile
      const long k0n = (long)(kt + 1) * KT;
      tdm_load_tile((unsigned)(uintptr_t)&As[cur ^ 1][0], A + (long)bm * K + k0n, K, BM);
      tdm_load_tile((unsigned)(uintptr_t)&Bs[cur ^ 1][0], W + (long)bn * K + k0n, K, BN);
    }

    // fragments per ISA 16-bit layout:
    // lanes 0-15:  K=[0..7] lo half, [16..23] hi half
    // lanes 16-31: K=[8..15] lo half, [24..31] hi half
    const bf16* as = &As[cur][0];
    const bf16* bs = &Bs[cur][0];
    v16bf af[4], bfm[2];
    #pragma unroll
    for (int i = 0; i < 4; ++i) {
      const int r0 = wm * 64 + i * 16;
      bf16x8 lo = *(const bf16x8*)&as[(r0 + lr) * LD + lh * 8];
      bf16x8 hi = *(const bf16x8*)&as[(r0 + lr) * LD + 16 + lh * 8];
      af[i] = __builtin_shufflevector(lo, hi, 0,1,2,3,4,5,6,7,8,9,10,11,12,13,14,15);
    }
    #pragma unroll
    for (int j = 0; j < 2; ++j) {
      const int r0 = wn * 32 + j * 16;
      bf16x8 lo = *(const bf16x8*)&bs[(r0 + lr) * LD + lh * 8];
      bf16x8 hi = *(const bf16x8*)&bs[(r0 + lr) * LD + 16 + lh * 8];
      bfm[j] = __builtin_shufflevector(lo, hi, 0,1,2,3,4,5,6,7,8,9,10,11,12,13,14,15);
    }

    #pragma unroll
    for (int i = 0; i < 4; ++i)
      #pragma unroll
      for (int j = 0; j < 2; ++j)
        c[i][j] = __builtin_amdgcn_wmma_f32_16x16x32_bf16(
            false, af[i], false, bfm[j], (short)0, c[i][j], false, false);
    // next-iteration barrier orders these LDS reads before buffer reuse
  }

  // ---- epilogue: C layout VGPR e = row (e + 8*lh), col lr ----
  #pragma unroll
  for (int i = 0; i < 4; ++i) {
    #pragma unroll
    for (int j = 0; j < 2; ++j) {
      const int gm0 = bm + wm * 64 + i * 16 + lh * 8;
      const int gn  = bn + wn * 32 + j * 16 + lr;
      const float bv = bias[gn];
      #pragma unroll
      for (int e = 0; e < 8; ++e) {
        const long gm = gm0 + e;
        float v = c[i][j][e] + bv;
        if (MODE == 0) {
          const int seg = gn / S6_DI;          // uniform within a block
          const int cn  = gn - seg * S6_DI;
          const long o  = gm * S6_DI + cn;
          if (seg == 0)      o_xp[o] = (bf16)(v * fast_sigmoid(v));
          else if (seg == 1) o_z[o]  = (bf16)(v * fast_sigmoid(v));
          else               o_dt[o] = (bf16)v;
        } else if (MODE == 1) {
          const float sp = (v > 20.0f) ? v : __logf(1.0f + __expf(v)); // v_log/v_exp
          outf[gm * (long)N + gn] = sp;
        } else {
          outf[gm * (long)N + gn] = v + resid[gm * (long)N + gn];
        }
      }
    }
  }
}

// ---------------- chunked selective scan (h resets every 32 steps) ----------
__global__ __launch_bounds__(256)
void scan_kernel(const bf16* __restrict__ xp, const float* __restrict__ dt,
                 const bf16* __restrict__ zb, const float* __restrict__ A_log,
                 const float* __restrict__ D_vec, bf16* __restrict__ yout) {
  const int t = blockIdx.x * 256 + threadIdx.x;       // (b,chunk,channel)
  if (t >= S6_B * S6_NC * S6_DI) return;
  const int d  = t % S6_DI;
  const int rc = t / S6_DI;                           // b*NC + c
  const long row0 = (long)rc * S6_CH;                 // = b*L + c*CH (exact)

  float Ac[S6_DS];
  #pragma unroll
  for (int s = 0; s < S6_DS; ++s) Ac[s] = -__expf(A_log[d * S6_DS + s]);
  const float Dd = D_vec[d];

  float h[S6_DS];
  #pragma unroll
  for (int s = 0; s < S6_DS; ++s) h[s] = 0.f;

  for (int step = 0; step < S6_CH; ++step) {
    const long idx = (row0 + step) * S6_DI + d;
    const float xv  = (float)xp[idx];
    const float dtv = dt[idx];
    const float zv  = (float)zb[idx];
    float y = 0.f;
    #pragma unroll
    for (int s = 0; s < S6_DS; ++s) {
      h[s] = h[s] * __expf(dtv * Ac[s]) + xv;         // v_exp_f32
      y += h[s];
    }
    yout[idx] = (bf16)(y * zv + xv * Dd);
  }
}

// ---------------- host launcher ----------------
extern "C" void kernel_launch(void* const* d_in, const int* in_sizes, int n_in,
                              void* d_out, int out_size, void* d_ws, size_t ws_size,
                              hipStream_t stream) {
  const float* x     = (const float*)d_in[0];
  const float* gamma = (const float*)d_in[1];
  const float* beta  = (const float*)d_in[2];
  const float* W_in  = (const float*)d_in[3];
  const float* b_in  = (const float*)d_in[4];
  const float* W_dt  = (const float*)d_in[5];
  const float* b_dt  = (const float*)d_in[6];
  const float* A_log = (const float*)d_in[7];
  const float* D_vec = (const float*)d_in[8];
  const float* W_out = (const float*)d_in[9];
  const float* b_out = (const float*)d_in[10];
  float* out = (float*)d_out;

  char* ws = (char*)d_ws;
  size_t off = 0;
  auto alloc = [&](size_t bytes) -> void* {
    off = (off + 255) & ~(size_t)255;
    void* p = ws + off;
    off += bytes;
    return p;
  };

  const long M = S6_M;
  bf16*  xn_b   = (bf16*)alloc((size_t)M * S6_DM * 2);
  bf16*  win_b  = (bf16*)alloc((size_t)3 * S6_DI * S6_DM * 2);
  bf16*  wdt_b  = (bf16*)alloc((size_t)S6_DI * S6_DI * 2);
  bf16*  wout_b = (bf16*)alloc((size_t)S6_DM * S6_DI * 2);
  bf16*  xp_b   = (bf16*)alloc((size_t)M * S6_DI * 2);
  bf16*  z_b    = (bf16*)alloc((size_t)M * S6_DI * 2);
  bf16*  dtin_b = (bf16*)alloc((size_t)M * S6_DI * 2);
  float* dt_f   = (float*)alloc((size_t)M * S6_DI * 4);
  bf16*  y_b    = (bf16*)alloc((size_t)M * S6_DI * 2);
  (void)ws_size; (void)in_sizes; (void)n_in; (void)out_size;

  // 1) weight conversions fp32 -> bf16
  {
    int n1 = 3 * S6_DI * S6_DM, n2 = S6_DI * S6_DI, n3 = S6_DM * S6_DI;
    cvt_bf16_kernel<<<(n1 + 255) / 256, 256, 0, stream>>>(W_in, win_b, n1);
    cvt_bf16_kernel<<<(n2 + 255) / 256, 256, 0, stream>>>(W_dt, wdt_b, n2);
    cvt_bf16_kernel<<<(n3 + 255) / 256, 256, 0, stream>>>(W_out, wout_b, n3);
  }

  // 2) LayerNorm -> xn bf16
  ln_kernel<<<(int)M, 256, 0, stream>>>(x, gamma, beta, xn_b);

  // 3) GEMM1: [M,256] @ [1152,256]^T -> silu/silu/raw split
  gemm_bf16_wmma<0><<<dim3(M / 128, (3 * S6_DI) / 128), 256, 0, stream>>>(
      xn_b, win_b, b_in, S6_DM, 3 * S6_DI,
      nullptr, xp_b, z_b, dtin_b, nullptr);

  // 4) GEMM2: [M,384] @ [384,384]^T -> softplus -> dt fp32
  gemm_bf16_wmma<1><<<dim3(M / 128, S6_DI / 128), 256, 0, stream>>>(
      dtin_b, wdt_b, b_dt, S6_DI, S6_DI,
      dt_f, nullptr, nullptr, nullptr, nullptr);

  // 5) chunked scan + gating -> y bf16
  {
    int nthr = S6_B * S6_NC * S6_DI;
    scan_kernel<<<(nthr + 255) / 256, 256, 0, stream>>>(
        xp_b, dt_f, z_b, A_log, D_vec, y_b);
  }

  // 6) GEMM3: [M,384] @ [256,384]^T + b_out + residual -> out fp32
  gemm_bf16_wmma<2><<<dim3(M / 128, S6_DM / 128), 256, 0, stream>>>(
      y_b, wout_b, b_out, S6_DI, S6_DM,
      out, nullptr, nullptr, nullptr, x);
}
